// ReservoirLinearRNN_Block_48576080118637
// MI455X (gfx1250) — compile-verified, matrix operand order checked by reading, not run on hardware
//
#include <hip/hip_runtime.h>
#include <hip/hip_bf16.h>

typedef __bf16 bf16_t;
typedef __attribute__((ext_vector_type(16))) __bf16 v16bf;
typedef __attribute__((ext_vector_type(8)))  float  v8f;

union ABFrag { v16bf v; uint4 q[2]; };

static __device__ __forceinline__ bf16_t f2bf(float f) {
    unsigned u = __float_as_uint(f);
    unsigned r = u + 0x7FFFu + ((u >> 16) & 1u);     // round-to-nearest-even
    unsigned short h = (unsigned short)(r >> 16);
    return __builtin_bit_cast(bf16_t, h);
}
static __device__ __forceinline__ float bf2f(bf16_t b) {
    unsigned short h = __builtin_bit_cast(unsigned short, b);
    return __uint_as_float(((unsigned)h) << 16);
}

// 16x32 bf16 operand fragment (A or B^T) from global, CDNA5 lane layout:
// lane group g = lane/16 owns K chunks [8g, 8g+8) and [16+8g, 16+8g+8)
static __device__ __forceinline__ v16bf load_frag(const bf16_t* __restrict__ base,
                                                  int row, int K, int k0, int g) {
    ABFrag f;
    const bf16_t* p = base + (size_t)row * K + k0 + (g << 3);
    f.q[0] = *(const uint4*)(p);
    f.q[1] = *(const uint4*)(p + 16);
    return f.v;
}
// Same fragment gather but from an LDS-resident row-major panel (emits ds_load_b128)
static __device__ __forceinline__ v16bf lds_frag(const bf16_t* l, int row, int K, int k0, int g) {
    ABFrag f;
    const bf16_t* p = l + row * K + k0 + (g << 3);
    f.q[0] = *(const uint4*)(p);
    f.q[1] = *(const uint4*)(p + 16);
    return f.v;
}

// Stage nelem bf16 (multiple of 8) from global to LDS across the whole block.
// Uses the gfx1250 async global->LDS DMA path (ASYNCcnt) via inline asm; the low 32
// bits of a generic pointer to __shared__ are the LDS byte offset (flat aperture rule).
static __device__ __forceinline__ void stage_to_lds(const bf16_t* __restrict__ g,
                                                    bf16_t* l, int nelem, int tid, int nthr) {
    int nvec = nelem >> 3;                 // 16-byte chunks
#if defined(__AMDGCN__)
    for (int i = tid; i < nvec; i += nthr) {
        unsigned long long ga = (unsigned long long)(const void*)(g + (size_t)i * 8);
        unsigned la = (unsigned)(unsigned long long)(const void*)(l + i * 8);
        asm volatile("global_load_async_to_lds_b128 %0, %1, off"
                     :: "v"(la), "v"(ga) : "memory");
    }
    asm volatile("s_wait_asynccnt 0x0" ::: "memory");
#else
    for (int i = tid; i < nvec; i += nthr)
        *(uint4*)(l + i * 8) = *(const uint4*)(g + (size_t)i * 8);
#endif
    __syncthreads();
}

// ---------------- conversions ----------------
__global__ void convert_f32_bf16(const float* __restrict__ in, bf16_t* __restrict__ out, int n) {
    int i = blockIdx.x * blockDim.x + threadIdx.x;
    if (i < n) out[i] = f2bf(in[i]);
}
__global__ void transpose_f32_to_bf16(const float* __restrict__ in, bf16_t* __restrict__ out,
                                      int R, int C) { // out[c*R+r] = in[r*C+c]
    int i = blockIdx.x * blockDim.x + threadIdx.x;
    if (i < R * C) { int r = i / C, c = i - r * C; out[(size_t)c * R + r] = f2bf(in[i]); }
}
__global__ void transpose_bf16(const bf16_t* __restrict__ in, bf16_t* __restrict__ out,
                               int R, int C) {
    int i = blockIdx.x * blockDim.x + threadIdx.x;
    if (i < R * C) { int r = i / C, c = i - r * C; out[(size_t)c * R + r] = in[i]; }
}

// ---------------- bf16 WMMA GEMM: C[M x N] = A[M x K] * Bt[N x K]^T (+bias) ----------------
// 256-thread block = 8 waves handles one 16-row M panel; A panel staged in LDS (async DMA),
// each wave register-blocks NTPW adjacent 16x16 N tiles (N == 8*NTPW*16).
template <int NTPW>
__global__ void gemm_bf16_panel(const bf16_t* __restrict__ A,
                                const bf16_t* __restrict__ Bt,
                                const float*  __restrict__ bias,
                                bf16_t* __restrict__ C,
                                int M, int N, int K, int permT) {
    __shared__ bf16_t lA[16 * 512];
    int tm   = blockIdx.x;
    int tid  = threadIdx.x;
    int wave = tid >> 5;
    int lane = tid & 31;
    int g = lane >> 4, mn = lane & 15;

    stage_to_lds(A + (size_t)tm * 16 * K, lA, 16 * K, tid, 256);

    int tnb = wave * NTPW;
    v8f acc[NTPW];
#pragma unroll
    for (int j = 0; j < NTPW; ++j) acc[j] = {};

    for (int k0 = 0; k0 < K; k0 += 32) {
        v16bf a = lds_frag(lA, mn, K, k0, g);
#pragma unroll
        for (int j = 0; j < NTPW; ++j) {
            v16bf b = load_frag(Bt, (tnb + j) * 16 + mn, K, k0, g);
            acc[j] = __builtin_amdgcn_wmma_f32_16x16x32_bf16(false, a, false, b, (short)0,
                                                             acc[j], false, false);
        }
    }
#pragma unroll
    for (int j = 0; j < NTPW; ++j) {
        int nn = (tnb + j) * 16 + mn;
        float bv = bias ? bias[nn] : 0.0f;
#pragma unroll
        for (int r = 0; r < 8; ++r) {
            int mm = tm * 16 + r + (g << 3);
            float v = acc[j][r] + bv;
            size_t orow;
            if (permT > 0) { int b_ = mm / permT; int t_ = mm - b_ * permT; orow = (size_t)t_ * 16 + b_; }
            else           { orow = (size_t)mm; }
            C[orow * (size_t)N + nn] = f2bf(v);
        }
    }
}

// ---------------- parallel linear-scan combine: out[i] = in[2i] @ P + in[2i+1] ----------------
// One block per pair: even vector (16x512 bf16, 16KB) staged in LDS, 8 waves x 4 tiles cover H.
__global__ void scan_combine(const bf16_t* __restrict__ Sin,
                             const bf16_t* __restrict__ Pt,
                             bf16_t* __restrict__ Sout,
                             int npairs) {
    const int H = 512;
    __shared__ bf16_t lA[16 * 512];
    int pair = blockIdx.x;
    int tid = threadIdx.x, wave = tid >> 5, lane = tid & 31;
    int g = lane >> 4, mn = lane & 15;
    const bf16_t* Aev = Sin + (size_t)(2 * pair)     * 16 * H;
    const bf16_t* Aod = Sin + (size_t)(2 * pair + 1) * 16 * H;

    stage_to_lds(Aev, lA, 16 * H, tid, 256);

    int tnb = wave * 4;
    v8f acc[4];
#pragma unroll
    for (int j = 0; j < 4; ++j) acc[j] = {};

    for (int k0 = 0; k0 < H; k0 += 32) {
        v16bf a = lds_frag(lA, mn, H, k0, g);
#pragma unroll
        for (int j = 0; j < 4; ++j) {
            v16bf b = load_frag(Pt, (tnb + j) * 16 + mn, H, k0, g);
            acc[j] = __builtin_amdgcn_wmma_f32_16x16x32_bf16(false, a, false, b, (short)0,
                                                             acc[j], false, false);
        }
    }
    bf16_t* Out = Sout + (size_t)pair * 16 * H;
#pragma unroll
    for (int j = 0; j < 4; ++j) {
        int nn = (tnb + j) * 16 + mn;
#pragma unroll
        for (int r = 0; r < 8; ++r) {
            int mm = r + (g << 3);
            float v = acc[j][r] + bf2f(Aod[(size_t)mm * H + nn]);
            Out[(size_t)mm * H + nn] = f2bf(v);
        }
    }
}

// ---------------- tail: z_last = h_final + x_last@Wr + br ; LayerNorm ----------------
__global__ void final_ln(const float* __restrict__ x, const bf16_t* __restrict__ hfin,
                         const float* __restrict__ Wr, const float* __restrict__ br,
                         const float* __restrict__ ln_g, const float* __restrict__ ln_b,
                         float* __restrict__ zn) {
    __shared__ float red[512];
    int row = blockIdx.x;        // 0..15
    int c   = threadIdx.x;       // 0..511
    const float* xl = x + ((size_t)row * 2048 + 2047) * 128;
    float s = bf2f(hfin[row * 512 + c]) + br[c];
    for (int k = 0; k < 128; ++k) s += xl[k] * Wr[k * 512 + c];
    red[c] = s; __syncthreads();
    for (int off = 256; off > 0; off >>= 1) { if (c < off) red[c] += red[c + off]; __syncthreads(); }
    float mu = red[0] * (1.0f / 512.0f); __syncthreads();
    float d = s - mu;
    red[c] = d * d; __syncthreads();
    for (int off = 256; off > 0; off >>= 1) { if (c < off) red[c] += red[c + off]; __syncthreads(); }
    float var = red[0] * (1.0f / 512.0f);
    zn[row * 512 + c] = d * rsqrtf(var + 1e-5f) * ln_g[c] + ln_b[c];
}

__global__ void final_mlp1(const float* __restrict__ zn, const float* __restrict__ W1,
                           const float* __restrict__ b1, float* __restrict__ t1) {
    int idx = blockIdx.x * blockDim.x + threadIdx.x;   // 16*1024
    if (idx >= 16 * 1024) return;
    int row = idx >> 10, col = idx & 1023;
    float s = b1[col];
    const float* z = zn + row * 512;
    for (int k = 0; k < 512; ++k) s += z[k] * W1[k * 1024 + col];
    t1[idx] = s > 0.0f ? s : 0.0f;
}
__global__ void final_mlp2(const float* __restrict__ t1, const float* __restrict__ W2,
                           const float* __restrict__ b2, float* __restrict__ out) {
    int idx = blockIdx.x * blockDim.x + threadIdx.x;   // 16*512
    if (idx >= 16 * 512) return;
    int row = idx >> 9, col = idx & 511;
    float s = b2[col];
    const float* t = t1 + row * 1024;
    for (int k = 0; k < 1024; ++k) s += t[k] * W2[k * 512 + col];
    out[idx] = s;
}

extern "C" void kernel_launch(void* const* d_in, const int* in_sizes, int n_in,
                              void* d_out, int out_size, void* d_ws, size_t ws_size,
                              hipStream_t stream) {
    (void)in_sizes; (void)n_in; (void)out_size; (void)ws_size;
    const float* x    = (const float*)d_in[0];
    const float* We   = (const float*)d_in[1];
    const float* be   = (const float*)d_in[2];
    const float* Wb   = (const float*)d_in[3];
    const float* Amat = (const float*)d_in[4];
    const float* Wr   = (const float*)d_in[5];
    const float* br   = (const float*)d_in[6];
    const float* ln_g = (const float*)d_in[7];
    const float* ln_b = (const float*)d_in[8];
    const float* W1   = (const float*)d_in[9];
    const float* b1   = (const float*)d_in[10];
    const float* W2   = (const float*)d_in[11];
    const float* b2   = (const float*)d_in[12];

    const int BT = 16 * 2048, L = 128, M = 256, H = 512;

    char* w = (char*)d_ws;
    size_t off = 0;
    auto alloc = [&](size_t bytes) { void* p = w + off; off += (bytes + 255) & ~(size_t)255; return p; };
    bf16_t* x_bf = (bf16_t*)alloc((size_t)BT * L * 2);
    bf16_t* WeT  = (bf16_t*)alloc((size_t)M * L * 2);
    bf16_t* WbT  = (bf16_t*)alloc((size_t)H * M * 2);
    bf16_t* E_bf = (bf16_t*)alloc((size_t)BT * M * 2);
    bf16_t* S0   = (bf16_t*)alloc((size_t)2048 * 16 * H * 2);  // Bu as (T,16,H)
    bf16_t* P0   = (bf16_t*)alloc((size_t)1024 * 16 * H * 2);
    bf16_t* P1   = (bf16_t*)alloc((size_t)512  * 16 * H * 2);
    bf16_t* Apw  = (bf16_t*)alloc((size_t)11 * H * H * 2);     // A^(2^k), row-major
    bf16_t* ApwT = (bf16_t*)alloc((size_t)11 * H * H * 2);     // transposed copies
    float*  zn   = (float*)alloc((size_t)16 * H * 4);
    float*  t1   = (float*)alloc((size_t)16 * 1024 * 4);

    // -- precision conversions / weight transposes (bf16, N x K layout) --
    convert_f32_bf16     <<<(BT * L + 255) / 256, 256, 0, stream>>>(x, x_bf, BT * L);
    transpose_f32_to_bf16<<<(L * M + 255) / 256, 256, 0, stream>>>(We, WeT, L, M);
    transpose_f32_to_bf16<<<(M * H + 255) / 256, 256, 0, stream>>>(Wb, WbT, M, H);
    convert_f32_bf16     <<<(H * H + 255) / 256, 256, 0, stream>>>(Amat, Apw, H * H);
    transpose_f32_to_bf16<<<(H * H + 255) / 256, 256, 0, stream>>>(Amat, ApwT, H, H);

    // -- E = x@We + be ; Bu = E@Wb (written permuted (T,16,H)) --
    gemm_bf16_panel<2><<<BT / 16, 256, 0, stream>>>(x_bf, WeT, be, E_bf, BT, M, L, 0);
    gemm_bf16_panel<4><<<BT / 16, 256, 0, stream>>>(E_bf, WbT, nullptr, S0, BT, H, M, 2048);

    // -- power chain: A^(2^k) by repeated squaring --
    for (int k = 1; k < 11; ++k) {
        bf16_t* Pp  = Apw  + (size_t)(k - 1) * H * H;
        bf16_t* PpT = ApwT + (size_t)(k - 1) * H * H;
        bf16_t* Pk  = Apw  + (size_t)k * H * H;
        bf16_t* PkT = ApwT + (size_t)k * H * H;
        gemm_bf16_panel<4><<<H / 16, 256, 0, stream>>>(Pp, PpT, nullptr, Pk, H, H, H, 0);
        transpose_bf16<<<(H * H + 255) / 256, 256, 0, stream>>>(Pk, PkT, H, H);
    }

    // -- log-depth parallel scan: h_{T-1} = sum_t bu_t A^{T-1-t} --
    const bf16_t* in = S0;
    for (int k = 0; k < 11; ++k) {
        int npairs = 1024 >> k;
        bf16_t* outb = (k & 1) ? P1 : P0;
        scan_combine<<<npairs, 256, 0, stream>>>(in, ApwT + (size_t)k * H * H, outb, npairs);
        in = outb;
    }
    const bf16_t* hfin = in;   // level 10 output lands in P0

    // -- tail: last-timestep LN + MLP --
    final_ln  <<<16, 512, 0, stream>>>(x, hfin, Wr, br, ln_g, ln_b, zn);
    final_mlp1<<<(16 * 1024) / 256, 256, 0, stream>>>(zn, W1, b1, t1);
    final_mlp2<<<(16 * 512)  / 256, 256, 0, stream>>>(t1, W2, b2, (float*)d_out);
}
